// EarthSpecificLayer_59339268161780
// MI455X (gfx1250) — compile-verified, matrix operand order checked by reading, not run on hardware
//
#include <hip/hip_runtime.h>
#include <hip/hip_bf16.h>

typedef __attribute__((ext_vector_type(16))) _Float16 v16h;
typedef __attribute__((ext_vector_type(8)))  float    v8f;

// ---------------- problem constants (hardcoded from reference) ----------------
constexpr int kWZ = 2, kWH = 6, kWW = 12;
constexpr int kWIN = 144;            // window tokens
constexpr int kNPAD = 160;           // padded attention inner dim (144 -> 160)
constexpr int kC  = 192;             // model dim
constexpr int kC3 = 576;             // 3*C
constexpr int kHEADS = 6, kHD = 32;
constexpr int kZ = 8, kH = 96, kW = 180;
constexpr int kL = kZ * kH * kW;     // 138240 tokens
constexpr int kWBZ = kZ / kWZ;       // 4
constexpr int kWBH = kH / kWH;       // 16
constexpr int kWBW = kW / kWW;       // 15
constexpr int kNW = kWBZ * kWBH * kWBW; // 960 windows
constexpr int kDEPTH = 2;
constexpr int kBIASROWS = (2 * kWW - 1) * kWH * kWH * kWZ * kWZ; // 3312
constexpr int kHID = 4 * kC;         // 768
constexpr float kNEG = -1.0e9f;
constexpr float kQSCALE = 0.17677669529663689f; // 1/sqrt(32)

// ---------------- WMMA fragment helpers (ISA 7.12.2 layouts) ----------------
// A 16x32 f16: lane l holds row (l&15); halves: K = 16*(h>>3) + 8*(l>=16) + (h&7)
__device__ __forceinline__ v16h frag_a_f16(const _Float16* base, int ld) {
  const _Float16* r = base + (threadIdx.x & 15) * ld + ((threadIdx.x & 16) ? 8 : 0);
  v16h a;
#pragma unroll
  for (int i = 0; i < 8; ++i) { a[i] = r[i]; a[8 + i] = r[16 + i]; }
  return a;
}
// packed B fragment: [tile][lane][16 halves] -> one contiguous 32B load per lane
__device__ __forceinline__ v16h frag_pk(const _Float16* pk, int tile) {
  return *((const v16h*)pk + (size_t)tile * 32 + (threadIdx.x & 31));
}
__device__ __forceinline__ v8f wmma32(v16h a, v16h b, v8f c) {
  return __builtin_amdgcn_wmma_f32_16x16x32_f16(false, a, false, b, (short)0, c, false, false);
}

// token index for (window w, window-local n) through the roll mapping
__device__ __forceinline__ int token_of(int w, int n, int shift) {
  int zb = w / (kWBH * kWBW);
  int hb = (w / kWBW) % kWBH;
  int wb = w % kWBW;
  int nz = n / (kWH * kWW);
  int nh = (n / kWW) % kWH;
  int nw = n % kWW;
  int z = zb * kWZ + nz, h = hb * kWH + nh, x = wb * kWW + nw;
  if (shift) { z = (z + kWZ / 2) % kZ; h = (h + kWH / 2) % kH; x = (x + kWW / 2) % kW; }
  return (z * kH + h) * kW + x;
}
// shift-mask region id for a window-local element (depends only on z,h)
__device__ __forceinline__ int region_of(int zb, int hb, int n) {
  int nz = n / (kWH * kWW);
  int nh = (n / kWW) % kWH;
  int z = zb * kWZ + nz, h = hb * kWH + nh;
  int rz = (z < kZ - kWZ) ? 0 : ((z < kZ - kWZ / 2) ? 1 : 2);
  int rh = (h < kH - kWH) ? 0 : ((h < kH - kWH / 2) ? 1 : 2);
  return rz * 3 + rh;
}

// ---------------- setup kernels ----------------
// pack a row-major KxN f32 weight into B-fragment order:
// dst[((kt*NT+nt)*32 + lane)*16 + h] = src[(kt*32 + 16*(h>>3) + 8*(lane>=16) + (h&7)) * N
//                                          + nt*16 + (lane&15)]
__global__ void k_pack(const float* __restrict__ s, _Float16* __restrict__ d,
                       int K, int N) {
  int idx = blockIdx.x * 256 + threadIdx.x;
  if (idx >= K * N) return;
  int h    = idx & 15;
  int lane = (idx >> 4) & 31;
  int tile = idx >> 9;
  int NT = N >> 4;
  int nt = tile % NT, kt = tile / NT;
  int kk = kt * 32 + 16 * (h >> 3) + ((lane & 16) ? 8 : 0) + (h & 7);
  int n  = nt * 16 + (lane & 15);
  d[idx] = (_Float16)s[(size_t)kk * N + n];
}

// expand relative-position bias: bias_full[head][n][m]
__global__ void k_bias(const float* __restrict__ bias_tab, float* __restrict__ bias_full,
                       int layer) {
  int idx = blockIdx.x * 256 + threadIdx.x;
  if (idx >= kWIN * kWIN) return;
  int n = idx / kWIN, m = idx % kWIN;
  int z1 = n / 72, h1 = (n / 12) % 6, w1 = n % 12;
  int z2 = m / 72, h2 = (m / 12) % 6, w2 = m % 12;
  int pos = (z1 + 2 * z2) * ((2 * kWW - 1) * kWH * kWH)
          + (h1 + kWH * h2) * (2 * kWW - 1)
          + (w1 - w2 + kWW - 1);
  const float* src = bias_tab + ((size_t)layer * kBIASROWS + pos) * kHEADS;
#pragma unroll
  for (int h = 0; h < kHEADS; ++h)
    bias_full[((size_t)h * kWIN + n) * kWIN + m] = src[h];
}

// ---------------- QKV projection (per window, gather + GEMM, f16 out) ----------------
__global__ __launch_bounds__(256) void k_qkv(const float* __restrict__ x,
                                             const _Float16* __restrict__ wq_pk,
                                             const float* __restrict__ qb,
                                             _Float16* __restrict__ qkv, int shift) {
  int w = blockIdx.x;
  int wave = threadIdx.x >> 5;
  int ll = threadIdx.x & 15;
  int hi = (threadIdx.x >> 4) & 1;
  _Float16* orow = qkv + (size_t)w * kWIN * kC3;
  int cur_mt = -1;
  v16h a[kC / 32];                     // cached A fragments for the current row block
  for (int t = wave; t < 9 * 36; t += 8) {   // 144/16 x 576/16 tiles
    int mt = t / 36, nt = t % 36;
    if (mt != cur_mt) {
      cur_mt = mt;
      int ta = token_of(w, mt * 16 + ll, shift);
      const float* arow = x + (size_t)ta * kC + (hi ? 8 : 0);
#pragma unroll
      for (int k = 0; k < kC / 32; ++k) {
        const float* r = arow + k * 32;
#pragma unroll
        for (int i = 0; i < 8; ++i) {
          a[k][i] = (_Float16)r[i];
          a[k][8 + i] = (_Float16)r[16 + i];
        }
      }
    }
    // issue all B loads as one clause, then run the WMMA chain
    v16h bq[kC / 32];
#pragma unroll
    for (int k = 0; k < kC / 32; ++k) bq[k] = frag_pk(wq_pk, k * 36 + nt);
    v8f acc = {};
#pragma unroll
    for (int k = 0; k < kC / 32; ++k) acc = wmma32(a[k], bq[k], acc);
    int col = nt * 16 + ll;
    float bias = qb[col];
    float scale = (col < kC) ? kQSCALE : 1.0f;  // fold q * hd^-0.5
#pragma unroll
    for (int r2 = 0; r2 < 8; ++r2) {
      int m = mt * 16 + r2 + hi * 8;
      orow[(size_t)m * kC3 + col] = (_Float16)((acc[r2] + bias) * scale);
    }
  }
}

// ---------------- windowed attention (1 wave = 1 head, flash-style row blocks) ------
__global__ __launch_bounds__(192) void k_attn(const _Float16* __restrict__ qkv,
                                              const float* __restrict__ bias_full,
                                              _Float16* __restrict__ attn_out, int shift) {
  extern __shared__ char smem[];
  _Float16* p_lds = (_Float16*)smem;                       // [6][16][160]
  _Float16* vpk   = (_Float16*)smem + kHEADS * 16 * kNPAD; // [6][5][2][32][16] packed V
  int w = blockIdx.x;
  int head = threadIdx.x >> 5;
  int ll = threadIdx.x & 15;
  int hi = (threadIdx.x >> 4) & 1;
  int zb = w / (kWBH * kWBW);
  int hb = (w / kWBW) % kWBH;
  const _Float16* qw = qkv + (size_t)w * kWIN * kC3;
  _Float16* prow = p_lds + head * 16 * kNPAD;
  const _Float16* vh = vpk + head * (5 * 2 * 512);

  // stage V into packed B-fragment layout (coalesced global reads, pad rows = 0)
  for (int idx = threadIdx.x; idx < kNPAD * kC; idx += 192) {
    int c  = idx % kC;     // contiguous across threads
    int kk = idx / kC;     // 0..159
    _Float16 val = (_Float16)0.f;
    if (kk < kWIN) val = qw[(size_t)kk * kC3 + 2 * kC + c];
    int hh = c >> 5, dd = c & 31;
    int kt = kk >> 5, sub = kk & 31;
    int lane = (dd & 15) + ((sub & 8) ? 16 : 0);
    int hlf  = (sub & 7) + ((sub & 16) ? 8 : 0);
    int dt = dd >> 4;
    vpk[((((hh * 5 + kt) << 1) + dt) << 9) + lane * 16 + hlf] = val;
  }
  // zero the K-padding columns of P
  for (int idx = threadIdx.x & 31; idx < 16 * (kNPAD - kWIN); idx += 32) {
    int rr = idx / (kNPAD - kWIN), cc = kWIN + idx % (kNPAD - kWIN);
    prow[rr * kNPAD + cc] = (_Float16)0.f;
  }

  // k^T fragments: invariant across all q blocks -> load once per window
  v16h bk[9];
#pragma unroll
  for (int kb = 0; kb < 9; ++kb)
    bk[kb] = frag_a_f16(qw + (size_t)(kb * 16) * kC3 + kC + head * kHD, kC3);

  __syncthreads();

  // packed-V fragments: invariant across q blocks -> load from LDS once
  v16h vf[10];
#pragma unroll
  for (int kt = 0; kt < kNPAD / 32; ++kt) {
    vf[kt * 2 + 0] = frag_pk(vh, kt * 2 + 0);
    vf[kt * 2 + 1] = frag_pk(vh, kt * 2 + 1);
  }

  for (int qb = 0; qb < 9; ++qb) {
    // S = q_block(16x32) @ k^T : 9 back-to-back WMMAs on resident fragments
    v16h aq = frag_a_f16(qw + (size_t)(qb * 16) * kC3 + head * kHD, kC3);
    v8f s[9];
#pragma unroll
    for (int kb = 0; kb < 9; ++kb) {
      v8f z = {};
      s[kb] = wmma32(aq, bk[kb], z);
    }
    // bias + shift mask + row max
    float rmax[8];
#pragma unroll
    for (int r = 0; r < 8; ++r) rmax[r] = -3.0e38f;
    int regn[8];
#pragma unroll
    for (int r = 0; r < 8; ++r) regn[r] = region_of(zb, hb, qb * 16 + r + hi * 8);
#pragma unroll
    for (int kb = 0; kb < 9; ++kb) {
      int m = kb * 16 + ll;
      int regm = region_of(zb, hb, m);
#pragma unroll
      for (int r = 0; r < 8; ++r) {
        int n = qb * 16 + r + hi * 8;
        float v = s[kb][r] + bias_full[((size_t)head * kWIN + n) * kWIN + m];
        if (shift && (regn[r] != regm)) v += kNEG;
        s[kb][r] = v;
        rmax[r] = fmaxf(rmax[r], v);
      }
    }
#pragma unroll
    for (int r = 0; r < 8; ++r)
      for (int msk = 8; msk >= 1; msk >>= 1)
        rmax[r] = fmaxf(rmax[r], __shfl_xor(rmax[r], msk, 32));
    float rsum[8];
#pragma unroll
    for (int r = 0; r < 8; ++r) rsum[r] = 0.f;
#pragma unroll
    for (int kb = 0; kb < 9; ++kb)
#pragma unroll
      for (int r = 0; r < 8; ++r) {
        float e = __expf(s[kb][r] - rmax[r]);
        s[kb][r] = e;
        rsum[r] += e;
      }
#pragma unroll
    for (int r = 0; r < 8; ++r) {
      for (int msk = 8; msk >= 1; msk >>= 1) rsum[r] += __shfl_xor(rsum[r], msk, 32);
      rsum[r] = 1.0f / rsum[r];
    }
    // P -> LDS (f16, row-major, padded) for A-fragment repack
#pragma unroll
    for (int kb = 0; kb < 9; ++kb)
#pragma unroll
      for (int r = 0; r < 8; ++r)
        prow[(r + hi * 8) * kNPAD + kb * 16 + ll] = (_Float16)(s[kb][r] * rsum[r]);
    __syncthreads();
    // O = P(16x160) @ V(160x32) with resident packed-V fragments
    v8f o0 = {}, o1 = {};
    v16h ap[kNPAD / 32];
#pragma unroll
    for (int kt = 0; kt < kNPAD / 32; ++kt) ap[kt] = frag_a_f16(prow + kt * 32, kNPAD);
#pragma unroll
    for (int kt = 0; kt < kNPAD / 32; ++kt) {
      o0 = wmma32(ap[kt], vf[kt * 2 + 0], o0);
      o1 = wmma32(ap[kt], vf[kt * 2 + 1], o1);
    }
#pragma unroll
    for (int r = 0; r < 8; ++r) {
      int n = qb * 16 + r + hi * 8;
      _Float16* orow = attn_out + ((size_t)w * kWIN + n) * kC + head * kHD;
      orow[ll] = (_Float16)o0[r];
      orow[16 + ll] = (_Float16)o1[r];
    }
    __syncthreads();
  }
}

// ------------- proj GEMM + fused LayerNorm + residual, scatter via inverse roll -----
__global__ __launch_bounds__(256) void k_proj(const _Float16* __restrict__ attn_in,
                                              const _Float16* __restrict__ wp_pk,
                                              const float* __restrict__ pb,
                                              const float* __restrict__ x_in,
                                              const float* __restrict__ ln_g,
                                              const float* __restrict__ ln_b,
                                              float* __restrict__ x_mid, int shift) {
  extern __shared__ char smem[];
  float (*res)[kC] = (float (*)[kC])smem;    // 144 x 192 f32 dynamic LDS
  int w = blockIdx.x;
  int wave = threadIdx.x >> 5;
  int ll = threadIdx.x & 15;
  int hi = (threadIdx.x >> 4) & 1;
  const _Float16* arow = attn_in + (size_t)w * kWIN * kC;
  for (int t = wave; t < 9 * 12; t += 8) {
    int mt = t / 12, nt = t % 12;
    v16h af[kC / 32], bf[kC / 32];
#pragma unroll
    for (int k = 0; k < kC / 32; ++k) {
      af[k] = frag_a_f16(arow + (size_t)(mt * 16) * kC + k * 32, kC);
      bf[k] = frag_pk(wp_pk, k * 12 + nt);
    }
    v8f acc = {};
#pragma unroll
    for (int k = 0; k < kC / 32; ++k) acc = wmma32(af[k], bf[k], acc);
    int col = nt * 16 + ll;
    float bias = pb[col];
#pragma unroll
    for (int r = 0; r < 8; ++r) res[mt * 16 + r + hi * 8][col] = acc[r] + bias;
  }
  __syncthreads();
  for (int n = threadIdx.x; n < kWIN; n += 256) {
    float mu = 0.f;
    for (int c = 0; c < kC; ++c) mu += res[n][c];
    mu *= (1.0f / kC);
    float var = 0.f;
    for (int c = 0; c < kC; ++c) { float d = res[n][c] - mu; var += d * d; }
    float inv = rsqrtf(var * (1.0f / kC) + 1e-5f);
    int tk = token_of(w, n, shift);
    const float* xi = x_in + (size_t)tk * kC;
    float* xo = x_mid + (size_t)tk * kC;
    for (int c = 0; c < kC; ++c)
      xo[c] = xi[c] + (res[n][c] - mu) * inv * ln_g[c] + ln_b[c];
  }
}

// ---------------- MLP1: h = gelu(x_mid @ w1 + b1) -> f16 -----------------------------
__global__ __launch_bounds__(256) void k_mlp1(const float* __restrict__ xm,
                                              const _Float16* __restrict__ w1_pk,
                                              const float* __restrict__ b1,
                                              _Float16* __restrict__ hbuf) {
  int m0 = blockIdx.x * 64, n0 = blockIdx.y * 128;
  int wave = threadIdx.x >> 5;
  int mw = wave >> 2, nw_ = wave & 3;      // 2x4 wave grid, 32x32 per wave
  int ll = threadIdx.x & 15;
  int hi = (threadIdx.x >> 4) & 1;
  int ntb = blockIdx.y * 8 + nw_ * 2;      // base packed-tile column (NT = 48)
  v8f acc[2][2] = {{{}, {}}, {{}, {}}};
  const float* ar0 = xm + (size_t)(m0 + mw * 32 + ll) * kC + (hi ? 8 : 0);
  const float* ar1 = xm + (size_t)(m0 + mw * 32 + 16 + ll) * kC + (hi ? 8 : 0);
#pragma unroll
  for (int k0 = 0; k0 < kC / 32; k0 += 3) {   // groups of 3 K-steps: clause 12 loads
    v16h a0[3], a1[3], b0[3], b1[3];
#pragma unroll
    for (int j = 0; j < 3; ++j) {
      int k = k0 + j;
#pragma unroll
      for (int i = 0; i < 8; ++i) {
        a0[j][i] = (_Float16)ar0[k * 32 + i];  a0[j][8 + i] = (_Float16)ar0[k * 32 + 16 + i];
        a1[j][i] = (_Float16)ar1[k * 32 + i];  a1[j][8 + i] = (_Float16)ar1[k * 32 + 16 + i];
      }
      b0[j] = frag_pk(w1_pk, k * 48 + ntb);
      b1[j] = frag_pk(w1_pk, k * 48 + ntb + 1);
    }
#pragma unroll
    for (int j = 0; j < 3; ++j) {
      acc[0][0] = wmma32(a0[j], b0[j], acc[0][0]);
      acc[0][1] = wmma32(a0[j], b1[j], acc[0][1]);
      acc[1][0] = wmma32(a1[j], b0[j], acc[1][0]);
      acc[1][1] = wmma32(a1[j], b1[j], acc[1][1]);
    }
  }
#pragma unroll
  for (int mi = 0; mi < 2; ++mi)
#pragma unroll
    for (int ni = 0; ni < 2; ++ni) {
      int col = n0 + nw_ * 32 + ni * 16 + ll;
      float bb = b1[col];
#pragma unroll
      for (int r = 0; r < 8; ++r) {
        int row = m0 + mw * 32 + mi * 16 + r + hi * 8;
        float v = acc[mi][ni][r] + bb;
        float g = 0.5f * v * (1.0f + erff(v * 0.70710678118f));  // exact gelu
        hbuf[(size_t)row * kHID + col] = (_Float16)g;
      }
    }
}

// ---------------- MLP2: x_out = x_mid + LN(h @ w2 + b2) ------------------------------
__global__ __launch_bounds__(256) void k_mlp2(const _Float16* __restrict__ hbuf,
                                              const _Float16* __restrict__ w2_pk,
                                              const float* __restrict__ b2,
                                              const float* __restrict__ xm,
                                              const float* __restrict__ ln_g,
                                              const float* __restrict__ ln_b,
                                              float* __restrict__ xout) {
  __shared__ float res[32][kC];             // 24KB
  int m0 = blockIdx.x * 32;
  int wave = threadIdx.x >> 5;
  int ll = threadIdx.x & 15;
  int hi = (threadIdx.x >> 4) & 1;
  for (int t = wave; t < 2 * 12; t += 8) {  // 32/16 x 192/16 tiles
    int mt = t / 12, nt = t % 12;
    const _Float16* arow = hbuf + (size_t)(m0 + mt * 16) * kHID;
    v8f acc = {};
#pragma unroll
    for (int k0 = 0; k0 < kHID / 32; k0 += 6) {  // groups of 6 K-steps
      v16h af[6], bf[6];
#pragma unroll
      for (int j = 0; j < 6; ++j) {
        af[j] = frag_a_f16(arow + (k0 + j) * 32, kHID);
        bf[j] = frag_pk(w2_pk, (k0 + j) * 12 + nt);
      }
#pragma unroll
      for (int j = 0; j < 6; ++j) acc = wmma32(af[j], bf[j], acc);
    }
    int col = nt * 16 + ll;
    float bb = b2[col];
#pragma unroll
    for (int r = 0; r < 8; ++r) res[mt * 16 + r + hi * 8][col] = acc[r] + bb;
  }
  __syncthreads();
  for (int n = threadIdx.x; n < 32; n += 256) {
    float mu = 0.f;
    for (int c = 0; c < kC; ++c) mu += res[n][c];
    mu *= (1.0f / kC);
    float var = 0.f;
    for (int c = 0; c < kC; ++c) { float d = res[n][c] - mu; var += d * d; }
    float inv = rsqrtf(var * (1.0f / kC) + 1e-5f);
    size_t row = (size_t)(m0 + n) * kC;
    for (int c = 0; c < kC; ++c)
      xout[row + c] = xm[row + c] + (res[n][c] - mu) * inv * ln_g[c] + ln_b[c];
  }
}

// ---------------- host orchestration ----------------
extern "C" void kernel_launch(void* const* d_in, const int* in_sizes, int n_in,
                              void* d_out, int out_size, void* d_ws, size_t ws_size,
                              hipStream_t stream) {
  const float* x       = (const float*)d_in[0];
  const float* qkv_w   = (const float*)d_in[1];
  const float* qkv_b   = (const float*)d_in[2];
  const float* proj_w  = (const float*)d_in[3];
  const float* proj_b  = (const float*)d_in[4];
  const float* biastab = (const float*)d_in[5];
  const float* ln1_g   = (const float*)d_in[6];
  const float* ln1_b   = (const float*)d_in[7];
  const float* ln2_g   = (const float*)d_in[8];
  const float* ln2_b   = (const float*)d_in[9];
  const float* mlp_w1  = (const float*)d_in[10];
  const float* mlp_b1  = (const float*)d_in[11];
  const float* mlp_w2  = (const float*)d_in[12];
  const float* mlp_b2  = (const float*)d_in[13];

  char* ws = (char*)d_ws;
  size_t off = 0;
  auto alloc = [&](size_t bytes) -> char* {
    off = (off + 255) & ~(size_t)255;
    char* p = ws + off;
    off += bytes;
    return p;
  };
  _Float16* wq16  = (_Float16*)alloc((size_t)kDEPTH * kC * kC3 * 2);
  _Float16* wp16  = (_Float16*)alloc((size_t)kDEPTH * kC * kC * 2);
  _Float16* w116  = (_Float16*)alloc((size_t)kDEPTH * kC * kHID * 2);
  _Float16* w216  = (_Float16*)alloc((size_t)kDEPTH * kHID * kC * 2);
  float*    biasf = (float*)   alloc((size_t)kHEADS * kWIN * kWIN * 4);
  _Float16* qkvb  = (_Float16*)alloc(((size_t)kNW * kWIN + 16) * kC3 * 2);
  _Float16* attnb = (_Float16*)alloc((size_t)kNW * kWIN * kC * 2);
  _Float16* hb    = (_Float16*)alloc((size_t)kL * kHID * 2);
  float*    xmid  = (float*)   alloc((size_t)kL * kC * 4);
  float*    xA    = (float*)   alloc((size_t)kL * kC * 4);

  // pack weights into B-fragment order (f32 -> f16), per layer
  auto pack = [&](const float* s, _Float16* d, int K, int N) {
    for (int l = 0; l < kDEPTH; ++l)
      k_pack<<<(K * N + 255) / 256, 256, 0, stream>>>(
          s + (size_t)l * K * N, d + (size_t)l * K * N, K, N);
  };
  pack(qkv_w,  wq16, kC, kC3);
  pack(proj_w, wp16, kC, kC);
  pack(mlp_w1, w116, kC, kHID);
  pack(mlp_w2, w216, kHID, kC);

  const size_t attn_lds = (size_t)(kHEADS * 16 * kNPAD + kHEADS * 5 * 2 * 512) * 2;
  const float* xin = x;
  for (int layer = 0; layer < kDEPTH; ++layer) {
    int shift = layer & 1;
    float* xout = (layer == kDEPTH - 1) ? (float*)d_out : xA;

    k_bias<<<(kWIN * kWIN + 255) / 256, 256, 0, stream>>>(biastab, biasf, layer);
    k_qkv<<<kNW, 256, 0, stream>>>(xin, wq16 + (size_t)layer * kC * kC3,
                                   qkv_b + (size_t)layer * kC3, qkvb, shift);
    k_attn<<<kNW, 192, attn_lds, stream>>>(qkvb, biasf, attnb, shift);
    k_proj<<<kNW, 256, kWIN * kC * sizeof(float), stream>>>(
        attnb, wp16 + (size_t)layer * kC * kC, proj_b + (size_t)layer * kC,
        xin, ln1_g + (size_t)layer * kC, ln1_b + (size_t)layer * kC, xmid, shift);
    k_mlp1<<<dim3(kL / 64, kHID / 128), 256, 0, stream>>>(
        xmid, w116 + (size_t)layer * kC * kHID, mlp_b1 + (size_t)layer * kHID, hb);
    k_mlp2<<<kL / 32, 256, 0, stream>>>(
        hb, w216 + (size_t)layer * kHID * kC, mlp_b2 + (size_t)layer * kC,
        xmid, ln2_g + (size_t)layer * kC, ln2_b + (size_t)layer * kC, xout);
    xin = xA;
  }
  (void)in_sizes; (void)n_in; (void)out_size; (void)ws_size;
}
